// LayerNormLSTMCell_10685878632881
// MI455X (gfx1250) — matrix operand order repaired and checked by reading, not compiled, not run on hardware
//
#include <hip/hip_runtime.h>
#include <hip/hip_bf16.h>
#include <math.h>

// ---------------------------------------------------------------------------
// LayerNorm LSTM cell for MI455X (gfx1250, wave32, WMMA).
// B=8192, I=H=1024, 4H=4096.
//   gi = input @ W_ih^T ; gh = hx @ W_hh^T           (bf16 WMMA, f32 accum)
//   gates = LN(gi) + LN(gh); i,f,g,o = split(gates)
//   c  = sigmoid(f)*cx + sigmoid(i)*tanh(g)
//   cy = LN_c(c); hy = sigmoid(o)*tanh(cy)
// Tile staging uses GLOBAL_LOAD_ASYNC_TO_LDS_B128 (ASYNCcnt) when available.
// ---------------------------------------------------------------------------

typedef __bf16 v8bf  __attribute__((ext_vector_type(8)));
typedef __bf16 v16bf __attribute__((ext_vector_type(16)));
typedef float  v8f   __attribute__((ext_vector_type(8)));
typedef int    v4i   __attribute__((ext_vector_type(4)));

typedef __attribute__((address_space(1))) v4i* as1_v4i_p;  // global
typedef __attribute__((address_space(3))) v4i* as3_v4i_p;  // LDS

#define K_DIM   1024
#define FOURH   4096
#define H_DIM   1024
#define BM      128
#define BN      128
#define BK      32
#define LDS_PAD 8            // row stride 40 bf16 = 80 B (16B aligned, bank-spread)
#define EPS     1e-5f

#if defined(__has_builtin)
#  if __has_builtin(__builtin_amdgcn_global_load_async_to_lds_b128)
#    define HAS_ASYNC_LDS 1
#  else
#    define HAS_ASYNC_LDS 0
#  endif
#  if __has_builtin(__builtin_amdgcn_s_wait_asynccnt)
#    define HAS_WAIT_ASYNC_BUILTIN 1
#  else
#    define HAS_WAIT_ASYNC_BUILTIN 0
#  endif
#else
#  define HAS_ASYNC_LDS 0
#  define HAS_WAIT_ASYNC_BUILTIN 0
#endif

// Pointer re-casts matching the compiler's addrspacecast bit semantics:
// AS1 pointer == 64-bit flat address; AS3 pointer == low 32 bits (LDS offset).
__device__ __forceinline__ as1_v4i_p to_global_v4i(const void* p) {
    return (as1_v4i_p)(unsigned long long)p;
}
__device__ __forceinline__ as3_v4i_p to_lds_v4i(void* p) {
    return (as3_v4i_p)(unsigned int)(unsigned long long)p;
}

// 16-byte global -> LDS copy; async path bypasses VGPR staging entirely.
__device__ __forceinline__ void copy16_g2l(const __bf16* __restrict__ g,
                                           __bf16* l) {
#if HAS_ASYNC_LDS
    __builtin_amdgcn_global_load_async_to_lds_b128(
        to_global_v4i(g), to_lds_v4i(l), /*offset=*/0, /*cpol=*/0);
#else
    *(v8bf*)l = *(const v8bf*)g;
#endif
}

__device__ __forceinline__ void drain_async() {
#if HAS_ASYNC_LDS
#  if HAS_WAIT_ASYNC_BUILTIN
    __builtin_amdgcn_s_wait_asynccnt(0);
#  else
    asm volatile("s_wait_asynccnt 0x0" ::: "memory");
#  endif
#endif
}

// ---------------------------------------------------------------------------
// fp32 -> bf16 conversion (grid-stride)
// ---------------------------------------------------------------------------
__global__ void cvt_f32_bf16(const float* __restrict__ src,
                             __bf16* __restrict__ dst, int n) {
    int i = blockIdx.x * blockDim.x + threadIdx.x;
    int stride = gridDim.x * blockDim.x;
    for (; i < n; i += stride) dst[i] = (__bf16)src[i];
}

// ---------------------------------------------------------------------------
// bf16 WMMA GEMM:  C[M,4096] = A[M,1024] * W[4096,1024]^T
// blockIdx.z selects (A0,W0)->C0 or (A1,W1)->C1.
// 256 threads = 8 waves; wave grid 4(M) x 2(N); each wave: 32x64 = 2x4 WMMA tiles.
// ---------------------------------------------------------------------------
__global__ __launch_bounds__(256)
void lstm_gemm_wmma(const __bf16* __restrict__ A0, const __bf16* __restrict__ A1,
                    const __bf16* __restrict__ W0, const __bf16* __restrict__ W1,
                    float* __restrict__ C0, float* __restrict__ C1) {
    const __bf16* __restrict__ A = blockIdx.z ? A1 : A0;
    const __bf16* __restrict__ W = blockIdx.z ? W1 : W0;
    float* __restrict__ C        = blockIdx.z ? C1 : C0;

    __shared__ __bf16 As[2][BM][BK + LDS_PAD];   // 2*128*40*2B = 20 KB
    __shared__ __bf16 Ws[2][BN][BK + LDS_PAD];   // 20 KB

    const int tid   = threadIdx.x;
    const int lane  = tid & 31;
    const int wave  = tid >> 5;       // 0..7
    const int waveM = wave & 3;       // 4 waves along M
    const int waveN = wave >> 2;      // 2 waves along N
    const int mW    = waveM * 32;
    const int nW    = waveN * 64;

    const int mBlk = blockIdx.y * BM;
    const int nBlk = blockIdx.x * BN;

    const int r  = lane & 15;         // row/col index within 16
    const int hi = lane >> 4;         // lane half selector

    v8f acc[2][4];
    #pragma unroll
    for (int i = 0; i < 2; ++i)
        #pragma unroll
        for (int j = 0; j < 4; ++j) acc[i][j] = (v8f)0.0f;

    const int NK = K_DIM / BK;        // 32 k-steps

    // ---- tile loader: 256 threads x (2+2) chunks of 8 bf16 (16 B) ----
    auto load_tiles = [&](int kb, int buf) {
        const int k0 = kb * BK;
        #pragma unroll
        for (int c = tid; c < BM * BK / 8; c += 256) {
            const int row = c >> 2, kc = c & 3;
            copy16_g2l(A + (size_t)(mBlk + row) * K_DIM + k0 + kc * 8,
                       &As[buf][row][kc * 8]);
        }
        #pragma unroll
        for (int c = tid; c < BN * BK / 8; c += 256) {
            const int row = c >> 2, kc = c & 3;
            copy16_g2l(W + (size_t)(nBlk + row) * K_DIM + k0 + kc * 8,
                       &Ws[buf][row][kc * 8]);
        }
    };

    load_tiles(0, 0);
    drain_async();
    __syncthreads();

    #pragma unroll 1
    for (int kb = 0; kb < NK; ++kb) {
        const int buf = kb & 1;
        if (kb + 1 < NK) load_tiles(kb + 1, buf ^ 1);   // overlaps with WMMAs

        // ---- A fragments (ISA 16-bit A 16x32 layout) ----
        // lanes 0-15: M=lane,   K = {0..7, 16..23}
        // lanes16-31: M=lane-16,K = {8..15,24..31}
        v16bf afrag[2];
        #pragma unroll
        for (int i = 0; i < 2; ++i) {
            v8bf lo = *(const v8bf*)&As[buf][mW + i * 16 + r][hi * 8];
            v8bf hh = *(const v8bf*)&As[buf][mW + i * 16 + r][16 + hi * 8];
            afrag[i] = __builtin_shufflevector(lo, hh, 0,1,2,3,4,5,6,7,
                                                       8,9,10,11,12,13,14,15);
        }
        // ---- B fragments: lane = N column, K = {0..7,16..23} / {8..15,24..31} per lane half ----
        v16bf bfrag[4];
        #pragma unroll
        for (int j = 0; j < 4; ++j) {
            v8bf lo = *(const v8bf*)&Ws[buf][nW + j * 16 + r][hi * 8];
            v8bf hh = *(const v8bf*)&Ws[buf][nW + j * 16 + r][16 + hi * 8];
            bfrag[j] = __builtin_shufflevector(lo, hh, 0,1,2,3,4,5,6,7,
                                                       8,9,10,11,12,13,14,15);
        }

        #pragma unroll
        for (int i = 0; i < 2; ++i)
            #pragma unroll
            for (int j = 0; j < 4; ++j)
                acc[i][j] = __builtin_amdgcn_wmma_f32_16x16x32_bf16(
                    /*neg_a=*/false, afrag[i], /*neg_b=*/false, bfrag[j],
                    /*c_mod=*/(short)0, acc[i][j],
                    /*reuse_a=*/false, /*reuse_b=*/false);

        drain_async();      // next-stage LDS writes complete before barrier
        __syncthreads();
    }

    // ---- store f32 C fragments (ISA 32-bit C/D layout) ----
    // VGPR rr: lanes 0-15 -> M=rr (N=lane); lanes 16-31 -> M=8+rr (N=lane-16)
    #pragma unroll
    for (int i = 0; i < 2; ++i) {
        #pragma unroll
        for (int j = 0; j < 4; ++j) {
            const int m0 = mBlk + mW + i * 16 + hi * 8;
            const int n0 = nBlk + nW + j * 16 + r;
            float* dst = C + (size_t)m0 * FOURH + n0;
            #pragma unroll
            for (int rr = 0; rr < 8; ++rr)
                dst[(size_t)rr * FOURH] = acc[i][j][rr];
        }
    }
}

// ---------------------------------------------------------------------------
// Fused epilogue: per batch-row LayerNorms + LSTM cell update.
// One 256-thread block per row b.
// ---------------------------------------------------------------------------
__device__ __forceinline__ float sigmoidf(float x) {
    return 1.0f / (1.0f + __expf(-x));
}

__global__ __launch_bounds__(256)
void ln_lstm_epilogue(const float* __restrict__ gi, const float* __restrict__ gh,
                      const float* __restrict__ cx,
                      const float* __restrict__ lniw, const float* __restrict__ lnib,
                      const float* __restrict__ lnhw, const float* __restrict__ lnhb,
                      const float* __restrict__ lncw, const float* __restrict__ lncb,
                      float* __restrict__ out, int Brows) {
    __shared__ float gates[FOURH];     // 16 KB
    __shared__ float4 red[256];        //  4 KB

    const int b = blockIdx.x;
    const int t = threadIdx.x;
    const float* gir = gi + (size_t)b * FOURH;
    const float* ghr = gh + (size_t)b * FOURH;

    // ---- pass 1: row stats of gi and gh (4096 each) ----
    float vgi[16], vgh[16];
    float si = 0.f, ssi = 0.f, sh = 0.f, ssh = 0.f;
    #pragma unroll
    for (int k = 0; k < 16; ++k) {
        const int j = t + k * 256;
        const float a = gir[j], c = ghr[j];
        vgi[k] = a; vgh[k] = c;
        si += a; ssi += a * a; sh += c; ssh += c * c;
    }
    red[t] = make_float4(si, ssi, sh, ssh);
    __syncthreads();
    #pragma unroll
    for (int s = 128; s > 0; s >>= 1) {
        if (t < s) {
            float4 o = red[t + s];
            red[t].x += o.x; red[t].y += o.y; red[t].z += o.z; red[t].w += o.w;
        }
        __syncthreads();
    }
    const float4 tot = red[0];
    const float mui = tot.x * (1.0f / FOURH);
    const float rsi = rsqrtf(tot.y * (1.0f / FOURH) - mui * mui + EPS);
    const float muh = tot.z * (1.0f / FOURH);
    const float rsh = rsqrtf(tot.w * (1.0f / FOURH) - muh * muh + EPS);
    __syncthreads();

    // ---- pass 2: gates = LN(gi) + LN(gh) ----
    #pragma unroll
    for (int k = 0; k < 16; ++k) {
        const int j = t + k * 256;
        gates[j] = lniw[j] * (vgi[k] - mui) * rsi + lnib[j]
                 + lnhw[j] * (vgh[k] - muh) * rsh + lnhb[j];
    }
    __syncthreads();

    // ---- pass 3: cell update + stats of c (1024) ----
    float cv[4];
    float sc = 0.f, ssc = 0.f;
    #pragma unroll
    for (int k = 0; k < 4; ++k) {
        const int h = t + k * 256;
        const float ig = sigmoidf(gates[h]);
        const float fg = sigmoidf(gates[h + H_DIM]);
        const float gg = tanhf(gates[h + 2 * H_DIM]);
        const float c  = fg * cx[(size_t)b * H_DIM + h] + ig * gg;
        cv[k] = c;
        sc += c; ssc += c * c;
    }
    red[t] = make_float4(sc, ssc, 0.f, 0.f);
    __syncthreads();
    #pragma unroll
    for (int s = 128; s > 0; s >>= 1) {
        if (t < s) { red[t].x += red[t + s].x; red[t].y += red[t + s].y; }
        __syncthreads();
    }
    const float muc = red[0].x * (1.0f / H_DIM);
    const float rsc = rsqrtf(red[0].y * (1.0f / H_DIM) - muc * muc + EPS);

    // ---- pass 4: cy = LN_c(c), hy = sigmoid(o)*tanh(cy) ----
    float* hy = out;
    float* cy = out + (size_t)Brows * H_DIM;
    #pragma unroll
    for (int k = 0; k < 4; ++k) {
        const int h  = t + k * 256;
        const float cyv = (cv[k] - muc) * rsc * lncw[h] + lncb[h];
        const float og  = sigmoidf(gates[h + 3 * H_DIM]);
        hy[(size_t)b * H_DIM + h] = og * tanhf(cyv);
        cy[(size_t)b * H_DIM + h] = cyv;
    }
}

// ---------------------------------------------------------------------------
// launch
// ---------------------------------------------------------------------------
extern "C" void kernel_launch(void* const* d_in, const int* in_sizes, int n_in,
                              void* d_out, int out_size, void* d_ws, size_t ws_size,
                              hipStream_t stream) {
    const float* input = (const float*)d_in[0];   // [B,1024]
    const float* hx    = (const float*)d_in[1];   // [B,1024]
    const float* cx    = (const float*)d_in[2];   // [B,1024]
    const float* wih   = (const float*)d_in[3];   // [4096,1024]
    const float* whh   = (const float*)d_in[4];   // [4096,1024]
    const float* lniw  = (const float*)d_in[5];
    const float* lnib  = (const float*)d_in[6];
    const float* lnhw  = (const float*)d_in[7];
    const float* lnhb  = (const float*)d_in[8];
    const float* lncw  = (const float*)d_in[9];
    const float* lncb  = (const float*)d_in[10];

    const int B = in_sizes[0] / K_DIM;            // 8192

    // workspace layout (256B aligned)
    char* ws = (char*)d_ws;
    auto align = [](size_t x) { return (x + 255) & ~(size_t)255; };
    size_t off = 0;
    __bf16* input_bf = (__bf16*)(ws + off); off = align(off + (size_t)B * K_DIM * 2);
    __bf16* hx_bf    = (__bf16*)(ws + off); off = align(off + (size_t)B * K_DIM * 2);
    __bf16* wih_bf   = (__bf16*)(ws + off); off = align(off + (size_t)FOURH * K_DIM * 2);
    __bf16* whh_bf   = (__bf16*)(ws + off); off = align(off + (size_t)FOURH * K_DIM * 2);
    float*  gi       = (float*)(ws + off);  off = align(off + (size_t)B * FOURH * 4);
    float*  gh       = (float*)(ws + off);  off = align(off + (size_t)B * FOURH * 4);

    // 1) fp32 -> bf16
    const int nAct = B * K_DIM, nWt = FOURH * K_DIM;
    cvt_f32_bf16<<<2048, 256, 0, stream>>>(input, input_bf, nAct);
    cvt_f32_bf16<<<2048, 256, 0, stream>>>(hx,    hx_bf,    nAct);
    cvt_f32_bf16<<<1024, 256, 0, stream>>>(wih,   wih_bf,   nWt);
    cvt_f32_bf16<<<1024, 256, 0, stream>>>(whh,   whh_bf,   nWt);

    // 2) both GEMMs (z=0: gi, z=1: gh)
    dim3 gg(FOURH / BN, B / BM, 2);
    lstm_gemm_wmma<<<gg, 256, 0, stream>>>(input_bf, hx_bf, wih_bf, whh_bf, gi, gh);

    // 3) fused LayerNorms + LSTM cell
    ln_lstm_epilogue<<<B, 256, 0, stream>>>(gi, gh, cx, lniw, lnib, lnhw, lnhb,
                                            lncw, lncb, (float*)d_out, B);
}
